// AllocationNet_75161927680743
// MI455X (gfx1250) — matrix operand order repaired and checked by reading, not compiled, not run on hardware
//
#include <hip/hip_runtime.h>
#include <hip/hip_bf16.h>

// ---------------------------------------------------------------------------
// Types for CDNA5 WMMA (wave32): v_wmma_f32_16x16x32_f16
// ---------------------------------------------------------------------------
typedef _Float16 f16;
typedef _Float16 v16h __attribute__((ext_vector_type(16)));
typedef _Float16 v8h  __attribute__((ext_vector_type(8)));
typedef float    v8f  __attribute__((ext_vector_type(8)));

__device__ __forceinline__ v16h mk16(v8h lo, v8h hi) {
  v16h r;
#pragma unroll
  for (int i = 0; i < 8; ++i) { r[i] = lo[i]; r[i + 8] = hi[i]; }
  return r;
}

// ---------------------------------------------------------------------------
// Weight convert + transpose: W (K x N) f32 -> Wt (N x K) f16
// Makes every WMMA B-fragment a single contiguous 32B load.
// ---------------------------------------------------------------------------
__global__ void k_wT(const float* __restrict__ W, f16* __restrict__ Wt, int K, int N) {
  int id = blockIdx.x * 256 + threadIdx.x;
  if (id >= K * N) return;
  int k = id / N, n = id - k * N;
  Wt[(size_t)n * K + k] = (f16)W[id];
}

// ---------------------------------------------------------------------------
// Generic GEMM:  C(MxN,f16) = act( A(MxK,f16) @ Wt^T + bias )
//   Wt is N x K row-major (pre-transposed).
//   Row remap on A: arow = (m/subRows)*fullRows + m%subRows  (sub==full==1 => id)
// Block: 256 threads = 8 waves; block tile 64x64; wave tile 16x32 (2 WMMAs).
// All M multiples of 64, N multiples of 64, K multiples of 32 in this net.
// ---------------------------------------------------------------------------
__global__ __launch_bounds__(256) void k_gemm(
    const f16* __restrict__ A, const f16* __restrict__ Wt,
    const float* __restrict__ bias, f16* __restrict__ C,
    int M, int K, int N, int subRows, int fullRows, int relu)
{
  const int lane = threadIdx.x & 31;
  const int wave = threadIdx.x >> 5;
  const int wm = wave & 3, wn = wave >> 2;
  const int m0 = blockIdx.y * 64 + wm * 16;
  const int n0 = blockIdx.x * 64 + wn * 32;

  // A fragment addressing (contiguous k order used here, same as B):
  //  lanes 0-15 : M=lane,   K in [0,16)
  //  lanes 16-31: M=lane-16,K in [16,32)
  const int lm   = m0 + (lane & 15);
  const int arow = (lm / subRows) * fullRows + (lm % subRows);
  const f16* Arow = A + (size_t)arow * K;
  const int aoff = (lane & 16) ? 16 : 0;

  // B fragment addressing (contiguous k order used here, same as A):
  //  lanes 0-15 : N=lane,    K in [0,16)
  //  lanes 16-31: N=lane-16, K in [16,32)   -> 16 contiguous halves in Wt row
  const int ncol0 = n0 + (lane & 15);
  const f16* B0 = Wt + (size_t)ncol0 * K + ((lane & 16) ? 16 : 0);
  const f16* B1 = B0 + (size_t)16 * K;

  v8f acc0 = {}; v8f acc1 = {};
  for (int k0 = 0; k0 < K; k0 += 32) {
    __builtin_prefetch(Arow + k0 + 128, 0, 0);   // global_prefetch_b8
    v8h alo = *reinterpret_cast<const v8h*>(Arow + k0 + aoff);
    v8h ahi = *reinterpret_cast<const v8h*>(Arow + k0 + aoff + 8);
    v16h a = mk16(alo, ahi);
    v16h b0 = *reinterpret_cast<const v16h*>(B0 + k0);
    v16h b1 = *reinterpret_cast<const v16h*>(B1 + k0);
    acc0 = __builtin_amdgcn_wmma_f32_16x16x32_f16(false, a, false, b0, (short)0, acc0, false, false);
    acc1 = __builtin_amdgcn_wmma_f32_16x16x32_f16(false, a, false, b1, (short)0, acc1, false, false);
  }

  // C/D layout: VGPR r: lanes 0-15 -> M=r, lanes 16-31 -> M=r+8; N = lane&15
  const int rbase = (lane & 16) ? 8 : 0;
  const float bv0 = bias ? bias[ncol0]      : 0.f;
  const float bv1 = bias ? bias[ncol0 + 16] : 0.f;
#pragma unroll
  for (int r = 0; r < 8; ++r) {
    int row = m0 + rbase + r;
    float v0 = acc0[r] + bv0;
    float v1 = acc1[r] + bv1;
    if (relu) { v0 = v0 > 0.f ? v0 : 0.f; v1 = v1 > 0.f ? v1 : 0.f; }
    C[(size_t)row * N + ncol0]      = (f16)v0;
    C[(size_t)row * N + ncol0 + 16] = (f16)v1;
  }
}

// ---------------------------------------------------------------------------
// Attention: one wave (32 threads) per (sequence, head, 16-row query block).
// qkv rows are 768 wide: [q | k | v], head h occupies cols h*32..h*32+31.
// S = (Q Kt)/sqrt(32) via WMMA; wave softmax; Z = P V via WMMA (keys padded
// to a multiple of 32 with zeros so no NaN leaks into the MAC tree).
// ---------------------------------------------------------------------------
__global__ __launch_bounds__(32) void k_attn(
    const f16* __restrict__ qkv, f16* __restrict__ out, int Ntok, int Npad)
{
  extern __shared__ char smem[];
  f16*   Kl = (f16*)smem;              // Npad*32 halves
  f16*   Vl = Kl + (size_t)Npad * 32;  // Npad*32 halves
  float* S  = (float*)(Vl + (size_t)Npad * 32); // 16*Npad floats
  f16*   P  = (f16*)(S + (size_t)16 * Npad);    // 16*Npad halves

  const int lane = threadIdx.x;
  const int nrb  = Ntok / 16;
  const int rb   = blockIdx.x % nrb;
  const int h    = (blockIdx.x / nrb) & 7;
  const int s    = blockIdx.x / (nrb * 8);
  const size_t rowbase = (size_t)s * Ntok;

  // Stage K and V for this head into LDS (zero pad beyond Ntok).
  for (int idx = lane; idx < Npad * 32; idx += 32) {
    int t = idx >> 5, d = idx & 31;
    f16 kv = (f16)0.f, vv = (f16)0.f;
    if (t < Ntok) {
      kv = qkv[(rowbase + t) * 768 + 256 + h * 32 + d];
      vv = qkv[(rowbase + t) * 768 + 512 + h * 32 + d];
    }
    Kl[idx] = kv; Vl[idx] = vv;
  }
  __syncthreads();

  // Q fragment (A layout) straight from global (contiguous per row).
  const int qrow = rb * 16 + (lane & 15);
  const f16* Qrow = qkv + (rowbase + qrow) * 768 + h * 32;
  const int aoff  = (lane & 16) ? 16 : 0;
  v16h aq = mk16(*reinterpret_cast<const v8h*>(Qrow + aoff),
                 *reinterpret_cast<const v8h*>(Qrow + aoff + 8));

  const float scale = 0.17677669529663687f;  // 1/sqrt(32)
  const int ncol  = lane & 15;
  const int koffB = (lane & 16) ? 16 : 0;
  const int rbase = (lane & 16) ? 8 : 0;

  // S tiles: one WMMA per 16 keys (K-dim = DK = 32, exact fit).
  for (int j = 0; j < nrb; ++j) {
    v16h bk = *reinterpret_cast<const v16h*>(Kl + (size_t)(j * 16 + ncol) * 32 + koffB);
    v8f c = {};
    c = __builtin_amdgcn_wmma_f32_16x16x32_f16(false, aq, false, bk, (short)0, c, false, false);
#pragma unroll
    for (int r = 0; r < 8; ++r)
      S[(size_t)(rbase + r) * Npad + j * 16 + ncol] = c[r] * scale;
  }
  __syncthreads();

  // Row softmax (rows striped across the wave).
  for (int r = 0; r < 16; ++r) {
    float mx = -1e30f;
    for (int c = lane; c < Ntok; c += 32) mx = fmaxf(mx, S[(size_t)r * Npad + c]);
    for (int o = 16; o > 0; o >>= 1) mx = fmaxf(mx, __shfl_xor(mx, o, 32));
    float sum = 0.f;
    for (int c = lane; c < Ntok; c += 32) {
      float e = __expf(S[(size_t)r * Npad + c] - mx);
      S[(size_t)r * Npad + c] = e; sum += e;
    }
    for (int o = 16; o > 0; o >>= 1) sum += __shfl_xor(sum, o, 32);
    float inv = 1.f / sum;
    for (int c = lane; c < Npad; c += 32)
      P[(size_t)r * Npad + c] = (c < Ntok) ? (f16)(S[(size_t)r * Npad + c] * inv) : (f16)0.f;
  }
  __syncthreads();

  // Z = P @ V, split dk into two 16-wide halves; accumulate over key blocks.
  for (int nh = 0; nh < 2; ++nh) {
    v8f acc = {};
    for (int kb = 0; kb < Npad / 32; ++kb) {
      const f16* Prow = P + (size_t)(lane & 15) * Npad + kb * 32;
      v16h pa = mk16(*reinterpret_cast<const v8h*>(Prow + aoff),
                     *reinterpret_cast<const v8h*>(Prow + aoff + 8));
      v16h bv;
      const int dk = nh * 16 + ncol;
#pragma unroll
      for (int j = 0; j < 16; ++j)
        bv[j] = Vl[(size_t)(kb * 32 + koffB + j) * 32 + dk];
      acc = __builtin_amdgcn_wmma_f32_16x16x32_f16(false, pa, false, bv, (short)0, acc, false, false);
    }
#pragma unroll
    for (int r = 0; r < 8; ++r)
      out[(rowbase + rb * 16 + rbase + r) * 256 + h * 32 + nh * 16 + ncol] = (f16)acc[r];
  }
}

// ---------------------------------------------------------------------------
// LayerNorm with residual: out = LN(x + y) * g + b   (in-place on x is safe)
// One 256-thread block per row (E = 256).
// ---------------------------------------------------------------------------
__global__ __launch_bounds__(256) void k_ln(
    const f16* __restrict__ x, const f16* __restrict__ y,
    const float* __restrict__ g, const float* __restrict__ b,
    f16* __restrict__ out)
{
  const int row = blockIdx.x, t = threadIdx.x;
  float v = (float)x[(size_t)row * 256 + t] + (float)y[(size_t)row * 256 + t];
  float s1 = v, s2 = v * v;
  for (int o = 16; o > 0; o >>= 1) { s1 += __shfl_xor(s1, o, 32); s2 += __shfl_xor(s2, o, 32); }
  __shared__ float rs1[8], rs2[8];
  if ((t & 31) == 0) { rs1[t >> 5] = s1; rs2[t >> 5] = s2; }
  __syncthreads();
  float S1 = 0.f, S2 = 0.f;
#pragma unroll
  for (int i = 0; i < 8; ++i) { S1 += rs1[i]; S2 += rs2[i]; }
  float mean = S1 * (1.f / 256.f);
  float var  = S2 * (1.f / 256.f) - mean * mean;
  float rstd = rsqrtf(var + 1e-5f);
  out[(size_t)row * 256 + t] = (f16)((v - mean) * rstd * g[t] + b[t]);
}

// ---------------------------------------------------------------------------
// Embeddings / pooling / mean
// ---------------------------------------------------------------------------
__global__ __launch_bounds__(256) void k_embed_rt(
    const float* __restrict__ xr, const float* __restrict__ xt,
    const float* __restrict__ w, const float* __restrict__ bb,
    f16* __restrict__ xenc)
{
  const int row = blockIdx.x;           // b*128 + n
  const int b = row >> 7, n = row & 127;
  const int e = threadIdx.x;
  float c0, c1, c2;
  if (n < 8) { const float* p = xr + ((size_t)b * 8 + n) * 3;        c0 = p[0]; c1 = p[1]; c2 = p[2]; }
  else       { const float* p = xt + ((size_t)b * 120 + (n - 8)) * 3; c0 = p[0]; c1 = p[1]; c2 = p[2]; }
  float v = c0 * w[e] + c1 * w[256 + e] + c2 * w[512 + e] + bb[e];
  xenc[((size_t)b * 160 + n) * 256 + e] = (f16)v;
}

__global__ __launch_bounds__(256) void k_embed_ob(
    const float* __restrict__ xob, const float* __restrict__ w,
    const float* __restrict__ bb, f16* __restrict__ out)
{
  const int row = blockIdx.x;           // b*512 + ob*16 + p, 131072 rows
  const int e = threadIdx.x;
  const float* p = xob + (size_t)row * 2;
  float v = p[0] * w[e] + p[1] * w[256 + e] + bb[e];
  out[(size_t)row * 256 + e] = (f16)v;
}

__global__ __launch_bounds__(256) void k_pool(
    const f16* __restrict__ locx, f16* __restrict__ xenc)
{
  const int g = blockIdx.x;             // b*32 + ob, 8192 groups
  const int e = threadIdx.x;
  float s = 0.f;
#pragma unroll
  for (int p = 0; p < 16; ++p) s += (float)locx[((size_t)g * 16 + p) * 256 + e];
  xenc[((size_t)(g >> 5) * 160 + 128 + (g & 31)) * 256 + e] = (f16)(s * (1.f / 16.f));
}

__global__ __launch_bounds__(256) void k_ave(
    const f16* __restrict__ xenc, float* __restrict__ ave)
{
  const int b = blockIdx.x, e = threadIdx.x;
  float s = 0.f;
  for (int n = 0; n < 128; ++n) s += (float)xenc[((size_t)b * 160 + n) * 256 + e];
  ave[(size_t)b * 256 + e] = s * (1.f / 128.f);
}

// ---------------------------------------------------------------------------
// Serial pointer-network decode: one 256-thread block per batch element,
// 127 sequential steps, everything in LDS. Sampling uses a deterministic
// hash RNG (JAX's threefry is not reproduced here).
// ---------------------------------------------------------------------------
__global__ __launch_bounds__(256) void k_decode(
    const f16* __restrict__ xenc, const float* __restrict__ ave,
    const f16* __restrict__ kh, const f16* __restrict__ vh,
    const f16* __restrict__ kout,
    const float* __restrict__ wq,  const float* __restrict__ wqb,
    const float* __restrict__ ww,  const float* __restrict__ wwb,
    const float* __restrict__ owq, const float* __restrict__ owqb,
    const float* __restrict__ costmap, const int* __restrict__ is_train,
    float* __restrict__ out)
{
  const int b = blockIdx.x, t = threadIdx.x;
  __shared__ float qin[512], qh[256], sbuf[8 * 128], zbuf[256], z2[256], qo[256];
  __shared__ float lg[128], pp[128];
  __shared__ unsigned mask[4];
  __shared__ int cur;
  if (t == 0) cur = 0;
  if (t < 4) mask[t] = 0u;
  float dist = 0.f;
  const int train = is_train[0] != 0;
  __syncthreads();

  for (int i = 0; i < 127; ++i) {
    if (t == 0) mask[cur >> 5] |= (1u << (cur & 31));
    __syncthreads();
    const int idx = cur;

    qin[t]       = (float)xenc[((size_t)b * 160 + idx) * 256 + t];
    qin[256 + t] = ave[(size_t)b * 256 + t];
    __syncthreads();

    { float acc = wqb[t];                       // q = [now, ave] @ dc_wq + b
      for (int k = 0; k < 512; ++k) acc += qin[k] * wq[(size_t)k * 256 + t];
      qh[t] = acc; }
    __syncthreads();

    for (int j = 0; j < 4; ++j) {               // s[h][n]
      int hn = t * 4 + j, hI = hn >> 7, n = hn & 127;
      float acc = 0.f;
      for (int d = 0; d < 32; ++d)
        acc += qh[hI * 32 + d] * (float)kh[((size_t)b * 128 + n) * 256 + hI * 32 + d];
      acc *= 0.17677669529663687f;
      if ((mask[n >> 5] >> (n & 31)) & 1u) acc = -1e9f;
      sbuf[hI * 128 + n] = acc;
    }
    __syncthreads();

    { const int hI = t >> 5, ln = t & 31;       // softmax per head (8 warps)
      float mx = -1e30f;
      for (int n = ln; n < 128; n += 32) mx = fmaxf(mx, sbuf[hI * 128 + n]);
      for (int o = 16; o > 0; o >>= 1) mx = fmaxf(mx, __shfl_xor(mx, o, 32));
      float sm = 0.f;
      for (int n = ln; n < 128; n += 32) { float e = __expf(sbuf[hI * 128 + n] - mx); sbuf[hI * 128 + n] = e; sm += e; }
      for (int o = 16; o > 0; o >>= 1) sm += __shfl_xor(sm, o, 32);
      float inv = 1.f / sm;
      for (int n = ln; n < 128; n += 32) sbuf[hI * 128 + n] *= inv; }
    __syncthreads();

    { const int hI = t >> 5; float acc = 0.f;   // z = a @ vh
      for (int n = 0; n < 128; ++n)
        acc += sbuf[hI * 128 + n] * (float)vh[((size_t)b * 128 + n) * 256 + t];
      zbuf[t] = acc; }
    __syncthreads();

    { float acc = wwb[t];                       // z @ dc_w + b
      for (int k = 0; k < 256; ++k) acc += zbuf[k] * ww[(size_t)k * 256 + t];
      z2[t] = acc; }
    __syncthreads();
    { float acc = owqb[t];                      // qo = z2 @ out_wq + b
      for (int k = 0; k < 256; ++k) acc += z2[k] * owq[(size_t)k * 256 + t];
      qo[t] = acc; }
    __syncthreads();

    if (t < 128) {                              // logits
      float acc = 0.f;
      for (int e = 0; e < 256; ++e)
        acc += qo[e] * (float)kout[((size_t)b * 128 + t) * 256 + e];
      float l = tanhf(acc * 0.0625f) * 10.f;
      if ((mask[t >> 5] >> (t & 31)) & 1u) l = -1e9f;
      lg[t] = l;
    }
    __syncthreads();

    if (t < 32) {                               // softmax over 128
      float mx = -1e30f;
      for (int n = t; n < 128; n += 32) mx = fmaxf(mx, lg[n]);
      for (int o = 16; o > 0; o >>= 1) mx = fmaxf(mx, __shfl_xor(mx, o, 32));
      float sm = 0.f;
      for (int n = t; n < 128; n += 32) { float e = __expf(lg[n] - mx); pp[n] = e; sm += e; }
      for (int o = 16; o > 0; o >>= 1) sm += __shfl_xor(sm, o, 32);
      float inv = 1.f / sm;
      for (int n = t; n < 128; n += 32) pp[n] *= inv;
    }
    __syncthreads();

    if (t == 0) {
      int nidx = 0;
      if (train) {
        unsigned x = 1234u ^ (unsigned)(i * 0x9E3779B9u) ^ (unsigned)(b * 0x85EBCA6Bu);
        x ^= x >> 16; x *= 0x7FEB352Du; x ^= x >> 15; x *= 0x846CA68Bu; x ^= x >> 16;
        float u = (float)(x >> 8) * (1.f / 16777216.f);
        float csum = 0.f; nidx = 127;
        for (int n = 0; n < 128; ++n) { csum += pp[n]; if (u < csum) { nidx = n; break; } }
      } else {
        float best = -1.f;
        for (int n = 0; n < 128; ++n) if (pp[n] > best) { best = pp[n]; nidx = n; }
      }
      out[(size_t)b * 127 + i] = (float)nidx;
      out[(size_t)256 * 127 + (size_t)b * 127 + i] = pp[nidx];
      dist += costmap[((size_t)b * 128 + idx) * 128 + nidx];
      cur = nidx;
    }
    __syncthreads();
  }
  if (t == 0) out[(size_t)2 * 256 * 127 + b] = dist;
}

// ---------------------------------------------------------------------------
// Host orchestration
// ---------------------------------------------------------------------------
extern "C" void kernel_launch(void* const* d_in, const int* in_sizes, int n_in,
                              void* d_out, int out_size, void* d_ws, size_t ws_size,
                              hipStream_t stream)
{
  (void)in_sizes; (void)n_in; (void)out_size; (void)ws_size;
  const float* x_r       = (const float*)d_in[0];
  const float* x_t       = (const float*)d_in[1];
  const float* x_ob      = (const float*)d_in[2];
  const float* costmap   = (const float*)d_in[3];
  const float* emb_rt_w  = (const float*)d_in[4];
  const float* emb_rt_b  = (const float*)d_in[5];
  const float* emb_ob_w  = (const float*)d_in[6];
  const float* emb_ob_b  = (const float*)d_in[7];
  const float* loc_qkv_w = (const float*)d_in[8];
  const float* loc_qkv_b = (const float*)d_in[9];
  const float* loc_o_w   = (const float*)d_in[10];
  const float* loc_o_b   = (const float*)d_in[11];
  const float* loc_ln1_g = (const float*)d_in[12];
  const float* loc_ln1_b = (const float*)d_in[13];
  const float* loc_ff1_w = (const float*)d_in[14];
  const float* loc_ff1_b = (const float*)d_in[15];
  const float* loc_ff2_w = (const float*)d_in[16];
  const float* loc_ff2_b = (const float*)d_in[17];
  const float* loc_ln2_g = (const float*)d_in[18];
  const float* loc_ln2_b = (const float*)d_in[19];
  const float* enc_qkv_w = (const float*)d_in[20];
  const float* enc_qkv_b = (const float*)d_in[21];
  const float* enc_o_w   = (const float*)d_in[22];
  const float* enc_o_b   = (const float*)d_in[23];
  const float* enc_ln1_g = (const float*)d_in[24];
  const float* enc_ln1_b = (const float*)d_in[25];
  const float* enc_ff1_w = (const float*)d_in[26];
  const float* enc_ff1_b = (const float*)d_in[27];
  const float* enc_ff2_w = (const float*)d_in[28];
  const float* enc_ff2_b = (const float*)d_in[29];
  const float* enc_ln2_g = (const float*)d_in[30];
  const float* enc_ln2_b = (const float*)d_in[31];
  const float* dc_wq_w   = (const float*)d_in[32];
  const float* dc_wq_b   = (const float*)d_in[33];
  const float* dc_wk_w   = (const float*)d_in[34];
  const float* dc_wk_b   = (const float*)d_in[35];
  const float* dc_wv_w   = (const float*)d_in[36];
  const float* dc_wv_b   = (const float*)d_in[37];
  const float* dc_w_w    = (const float*)d_in[38];
  const float* dc_w_b    = (const float*)d_in[39];
  const float* out_wq_w  = (const float*)d_in[40];
  const float* out_wq_b  = (const float*)d_in[41];
  const float* out_wk_w  = (const float*)d_in[42];
  const float* out_wk_b  = (const float*)d_in[43];
  const int*   is_train  = (const int*)d_in[44];
  float* dout = (float*)d_out;

  char* base = (char*)d_ws;
  size_t cur = 0;
  auto alloc = [&](size_t bytes) -> void* {
    void* p = base + cur;
    cur += (bytes + 255) & ~(size_t)255;
    return p;
  };

  const size_t MLOC = 131072;  // B*OBN*OBP rows
  f16*   bufA  = (f16*)alloc(MLOC * 256 * 2);
  f16*   bufY  = (f16*)alloc(MLOC * 256 * 2);
  f16*   bufB  = (f16*)alloc(MLOC * 256 * 2);
  f16*   bufQH = (f16*)alloc(MLOC * 1024 * 2);
  f16*   xenc  = (f16*)alloc((size_t)40960 * 256 * 2);
  f16*   khb   = (f16*)alloc((size_t)32768 * 256 * 2);
  f16*   vhb   = (f16*)alloc((size_t)32768 * 256 * 2);
  f16*   koutb = (f16*)alloc((size_t)32768 * 256 * 2);
  float* aveb  = (float*)alloc((size_t)256 * 256 * 4);

  f16 *qkvT[5], *oT[5], *f1T[5], *f2T[5];
  for (int l = 0; l < 5; ++l) {
    qkvT[l] = (f16*)alloc((size_t)256 * 768 * 2);
    oT[l]   = (f16*)alloc((size_t)256 * 256 * 2);
    f1T[l]  = (f16*)alloc((size_t)256 * 1024 * 2);
    f2T[l]  = (f16*)alloc((size_t)1024 * 256 * 2);
  }
  f16* dkT = (f16*)alloc((size_t)256 * 256 * 2);
  f16* dvT = (f16*)alloc((size_t)256 * 256 * 2);
  f16* okT = (f16*)alloc((size_t)256 * 256 * 2);

  auto convT = [&](const float* W, f16* Wt, int K, int N) {
    int tot = K * N;
    k_wT<<<(tot + 255) / 256, 256, 0, stream>>>(W, Wt, K, N);
  };
  for (int l = 0; l < 2; ++l) {
    convT(loc_qkv_w + (size_t)l * 256 * 768, qkvT[l], 256, 768);
    convT(loc_o_w   + (size_t)l * 256 * 256, oT[l],   256, 256);
    convT(loc_ff1_w + (size_t)l * 256 * 1024, f1T[l], 256, 1024);
    convT(loc_ff2_w + (size_t)l * 1024 * 256, f2T[l], 1024, 256);
  }
  for (int l = 0; l < 3; ++l) {
    convT(enc_qkv_w + (size_t)l * 256 * 768, qkvT[2 + l], 256, 768);
    convT(enc_o_w   + (size_t)l * 256 * 256, oT[2 + l],   256, 256);
    convT(enc_ff1_w + (size_t)l * 256 * 1024, f1T[2 + l], 256, 1024);
    convT(enc_ff2_w + (size_t)l * 1024 * 256, f2T[2 + l], 1024, 256);
  }
  convT(dc_wk_w, dkT, 256, 256);
  convT(dc_wv_w, dvT, 256, 256);
  convT(out_wk_w, okT, 256, 256);

  auto gemm = [&](const f16* A, const f16* Wt, const float* bias, f16* C,
                  int M, int K, int N, int sub, int full, int relu) {
    dim3 g(N / 64, M / 64);
    k_gemm<<<g, 256, 0, stream>>>(A, Wt, bias, C, M, K, N, sub, full, relu);
  };
  auto attn = [&](const f16* qkv, f16* outp, int Ntok, int numSeq) {
    int Npad = (Ntok + 31) & ~31;
    int nrb = Ntok / 16;
    size_t shm = (size_t)Npad * 224;   // K/V (f16) + S (f32) + P (f16)
    k_attn<<<numSeq * 8 * nrb, 32, shm, stream>>>(qkv, outp, Ntok, Npad);
  };
  auto block = [&](f16* X, int Mrows, int Ntok, int numSeq,
                   f16* qkvTw, const float* qkvB, f16* oTw, const float* oB,
                   const float* g1, const float* b1, f16* f1Tw, const float* f1B,
                   f16* f2Tw, const float* f2B, const float* g2, const float* b2) {
    gemm(X, qkvTw, qkvB, bufQH, Mrows, 256, 768, 1, 1, 0);
    attn(bufQH, bufY, Ntok, numSeq);
    gemm(bufY, oTw, oB, bufB, Mrows, 256, 256, 1, 1, 0);
    k_ln<<<Mrows, 256, 0, stream>>>(X, bufB, g1, b1, X);
    gemm(X, f1Tw, f1B, bufQH, Mrows, 256, 1024, 1, 1, 1);
    gemm(bufQH, f2Tw, f2B, bufB, Mrows, 1024, 256, 1, 1, 0);
    k_ln<<<Mrows, 256, 0, stream>>>(X, bufB, g2, b2, X);
  };

  // Embeddings
  k_embed_ob<<<131072, 256, 0, stream>>>(x_ob, emb_ob_w, emb_ob_b, bufA);
  k_embed_rt<<<256 * 128, 256, 0, stream>>>(x_r, x_t, emb_rt_w, emb_rt_b, xenc);

  // Local transformer (8192 seqs x 16 tokens)
  for (int l = 0; l < 2; ++l)
    block(bufA, 131072, 16, 8192,
          qkvT[l], loc_qkv_b + (size_t)l * 768, oT[l], loc_o_b + (size_t)l * 256,
          loc_ln1_g + (size_t)l * 256, loc_ln1_b + (size_t)l * 256,
          f1T[l], loc_ff1_b + (size_t)l * 1024,
          f2T[l], loc_ff2_b + (size_t)l * 256,
          loc_ln2_g + (size_t)l * 256, loc_ln2_b + (size_t)l * 256);

  // Mean-pool obstacles into encoder sequence rows 128..159
  k_pool<<<8192, 256, 0, stream>>>(bufA, xenc);

  // Global encoder (256 seqs x 160 tokens)
  for (int l = 0; l < 3; ++l)
    block(xenc, 40960, 160, 256,
          qkvT[2 + l], enc_qkv_b + (size_t)l * 768, oT[2 + l], enc_o_b + (size_t)l * 256,
          enc_ln1_g + (size_t)l * 256, enc_ln1_b + (size_t)l * 256,
          f1T[2 + l], enc_ff1_b + (size_t)l * 1024,
          f2T[2 + l], enc_ff2_b + (size_t)l * 256,
          enc_ln2_g + (size_t)l * 256, enc_ln2_b + (size_t)l * 256);

  // Decoder precomputes over the first 128 of each 160-row group.
  gemm(xenc, dkT, dc_wk_b,  khb,   32768, 256, 256, 128, 160, 0);
  gemm(xenc, dvT, dc_wv_b,  vhb,   32768, 256, 256, 128, 160, 0);
  gemm(xenc, okT, out_wk_b, koutb, 32768, 256, 256, 128, 160, 0);
  k_ave<<<256, 256, 0, stream>>>(xenc, aveb);

  // Serial decode, one block per batch element.
  k_decode<<<256, 256, 0, stream>>>(xenc, aveb, khb, vhb, koutb,
                                    dc_wq_w, dc_wq_b, dc_w_w, dc_w_b,
                                    out_wq_w, out_wq_b, costmap, is_train, dout);
}
